// MagnitudeAwareEncoding_51917564674449
// MI455X (gfx1250) — matrix operand order repaired and checked by reading, not compiled, not run on hardware
//
#include <hip/hip_runtime.h>
#include <hip/hip_bf16.h>

typedef __attribute__((ext_vector_type(16))) _Float16 v16h;
typedef __attribute__((ext_vector_type(8)))  _Float16 v8h;
typedef __attribute__((ext_vector_type(8)))  float    v8f;
typedef unsigned int u32x4 __attribute__((ext_vector_type(4)));
typedef int          i32x8 __attribute__((ext_vector_type(8)));
typedef int          i32x4 __attribute__((ext_vector_type(4)));

#define B_N   16384
#define D_N   768
#define D4_N  192
#define TD_N  2304
#define NB2   20

__device__ __forceinline__ float gelu_exact(float v){
    return 0.5f * v * (1.0f + erff(v * 0.70710678118654752f));
}

// --- TDM 2D tile load: rows of `tile_d0` contiguous 2-byte elements,
//     `tile_d1` rows with row stride `stride_el` elements, into LDS at lds_off.
__device__ __forceinline__ void tdm_load_2d_f16(unsigned int lds_off,
                                                const void* gptr,
                                                unsigned int tile_d0,
                                                unsigned int tile_d1,
                                                unsigned long long stride_el)
{
    unsigned long long ga = (unsigned long long)(uintptr_t)gptr;
    const unsigned int td = 0x7FFFFFFFu;      // huge tensor dims: no OOB clamp
    u32x4 g0;
    g0[0] = 1u;                                // count=1, user-mode descriptor
    g0[1] = lds_off;                           // LDS byte address
    g0[2] = (unsigned int)(ga & 0xFFFFFFFFu);  // global_addr[31:0]
    g0[3] = (unsigned int)((ga >> 32) & 0x01FFFFFFu) | 0x80000000u; // addr[56:32] | type=2
    i32x8 g1;
    g1[0] = (int)(1u << 16);                                   // data_size = 2 bytes
    g1[1] = (int)((td & 0xFFFFu) << 16);                       // tensor_dim0 lo
    g1[2] = (int)((td >> 16) | ((td & 0xFFFFu) << 16));        // dim0 hi | dim1 lo
    g1[3] = (int)((td >> 16) | (tile_d0 << 16));               // dim1 hi | tile_dim0
    g1[4] = (int)(tile_d1 & 0xFFFFu);                          // tile_dim1 | tile_dim2=0
    g1[5] = (int)(stride_el & 0xFFFFFFFFull);                  // dim0_stride lo
    g1[6] = (int)((stride_el >> 32) & 0xFFFFull);              // dim0_stride hi
    g1[7] = 0;
    i32x4 gz4 = {0, 0, 0, 0};
    i32x8 gz8 = {0, 0, 0, 0, 0, 0, 0, 0};
    __builtin_amdgcn_tensor_load_to_lds(g0, g1, gz4, gz4, gz8, 0);
}

// ---------------- Kernel 0: global maxima -------------------------------
__global__ __launch_bounds__(256) void k_maxred(const float* __restrict__ num,
                                                float* __restrict__ maxv){
    __shared__ float s1[256], s2[256];
    float m1 = 0.f, m2 = 0.f;
    for (int i = threadIdx.x; i < B_N; i += 256){
        float a = fabsf(num[i]);
        m1 = fmaxf(m1, a);
        m2 = fmaxf(m2, log1pf(a));
    }
    s1[threadIdx.x] = m1; s2[threadIdx.x] = m2;
    __syncthreads();
    for (int off = 128; off > 0; off >>= 1){
        if (threadIdx.x < off){
            s1[threadIdx.x] = fmaxf(s1[threadIdx.x], s1[threadIdx.x + off]);
            s2[threadIdx.x] = fmaxf(s2[threadIdx.x], s2[threadIdx.x + off]);
        }
        __syncthreads();
    }
    if (threadIdx.x == 0){ maxv[0] = s1[0]; maxv[1] = s2[0]; }
}

// ---------------- Kernel W: f32 -> f16 weight conversion -----------------
__global__ __launch_bounds__(256) void k_cvt(const float* __restrict__ src,
                                             _Float16* __restrict__ dst, int n){
    int i = blockIdx.x * blockDim.x + threadIdx.x;
    int stride = gridDim.x * blockDim.x;
    for (; i < n; i += stride) dst[i] = (_Float16)src[i];
}

// ---------------- Kernel 1: per-element features ------------------------
__global__ __launch_bounds__(256) void k_features(
    const float* __restrict__ number, const float* __restrict__ boundaries,
    const float* __restrict__ mag_emb,
    const float* __restrict__ se_w1, const float* __restrict__ se_b1,
    const float* __restrict__ se_g1, const float* __restrict__ se_bt1,
    const float* __restrict__ se_w2, const float* __restrict__ se_b2,
    const float* __restrict__ ne_w,  const float* __restrict__ ne_b,
    const float* __restrict__ ne_g,  const float* __restrict__ ne_bt,
    const float* __restrict__ ce_w,  const float* __restrict__ ce_b,
    const float* __restrict__ ce_g,  const float* __restrict__ ce_bt,
    const float* __restrict__ mag_scale, const float* __restrict__ temperature,
    const float* __restrict__ maxv,
    _Float16* __restrict__ A1, float* __restrict__ sc)
{
    __shared__ float rs[256], rq[256];
    __shared__ float hbuf[D4_N];
    __shared__ float stat[2];

    const int b   = blockIdx.x;
    const int tid = threadIdx.x;
    const float x  = number[b];
    const float ax = fabsf(x);
    const float la = log1pf(ax);
    const float sgn = (x > 0.f) ? 1.f : ((x < 0.f) ? -1.f : 0.f);
    const float sf  = floorf(log10f(ax + 1e-10f));
    const float nf0 = x  / (maxv[0] + 1e-10f);
    const float nf1 = la / (maxv[1] + 1e-10f);
    const float cf0 = sgn;
    const float cf1 = sf * (1.f / 16.f);

    int bin = 0;
    for (int i = 0; i < NB2; ++i) bin += (boundaries[i] < x) ? 1 : 0;
    if (bin > NB2 - 1) bin = NB2 - 1;

    if (tid == 0){
        float z = mag_scale[bin] / temperature[0];
        sc[b] = (z > 20.f) ? z : log1pf(expf(z));   // softplus
    }

    _Float16* rowN = A1 + (size_t)(b * 4 + 0) * D_N;
    _Float16* rowM = A1 + (size_t)(b * 4 + 1) * D_N;
    _Float16* rowS = A1 + (size_t)(b * 4 + 2) * D_N;
    _Float16* rowC = A1 + (size_t)(b * 4 + 3) * D_N;

    const float* me = mag_emb + (size_t)bin * D_N;
    for (int d = tid; d < D_N; d += 256) rowM[d] = (_Float16)me[d];

    float y[3];

    // ---- num_enc ----
    float ps = 0.f, pq = 0.f;
    #pragma unroll
    for (int i = 0; i < 3; ++i){
        int d = tid + i * 256;
        float v = nf0 * ne_w[2 * d] + nf1 * ne_w[2 * d + 1] + ne_b[d];
        y[i] = v; ps += v; pq += v * v;
    }
    rs[tid] = ps; rq[tid] = pq;
    __syncthreads();
    for (int off = 128; off > 0; off >>= 1){
        if (tid < off){ rs[tid] += rs[tid + off]; rq[tid] += rq[tid + off]; }
        __syncthreads();
    }
    if (tid == 0){ float m = rs[0] / (float)D_N; stat[0] = m; stat[1] = rq[0] / (float)D_N - m * m; }
    __syncthreads();
    {
        float m = stat[0], inv = rsqrtf(stat[1] + 1e-5f);
        #pragma unroll
        for (int i = 0; i < 3; ++i){
            int d = tid + i * 256;
            float v = (y[i] - m) * inv * ne_g[d] + ne_bt[d];
            rowN[d] = (_Float16)gelu_exact(v);
        }
    }
    __syncthreads();

    // ---- ctx_enc ----
    ps = 0.f; pq = 0.f;
    #pragma unroll
    for (int i = 0; i < 3; ++i){
        int d = tid + i * 256;
        float v = cf0 * ce_w[2 * d] + cf1 * ce_w[2 * d + 1] + ce_b[d];
        y[i] = v; ps += v; pq += v * v;
    }
    rs[tid] = ps; rq[tid] = pq;
    __syncthreads();
    for (int off = 128; off > 0; off >>= 1){
        if (tid < off){ rs[tid] += rs[tid + off]; rq[tid] += rq[tid + off]; }
        __syncthreads();
    }
    if (tid == 0){ float m = rs[0] / (float)D_N; stat[0] = m; stat[1] = rq[0] / (float)D_N - m * m; }
    __syncthreads();
    {
        float m = stat[0], inv = rsqrtf(stat[1] + 1e-5f);
        #pragma unroll
        for (int i = 0; i < 3; ++i){
            int d = tid + i * 256;
            float v = (y[i] - m) * inv * ce_g[d] + ce_bt[d];
            rowC[d] = (_Float16)gelu_exact(v);
        }
    }
    __syncthreads();

    // ---- scale path ----
    float tv = 0.f; ps = 0.f; pq = 0.f;
    if (tid < D4_N){
        tv = sf * se_w1[tid] + se_b1[tid];
        ps = tv; pq = tv * tv;
    }
    rs[tid] = ps; rq[tid] = pq;
    __syncthreads();
    for (int off = 128; off > 0; off >>= 1){
        if (tid < off){ rs[tid] += rs[tid + off]; rq[tid] += rq[tid + off]; }
        __syncthreads();
    }
    if (tid == 0){ float m = rs[0] / (float)D4_N; stat[0] = m; stat[1] = rq[0] / (float)D4_N - m * m; }
    __syncthreads();
    if (tid < D4_N){
        float v = (tv - stat[0]) * rsqrtf(stat[1] + 1e-5f) * se_g1[tid] + se_bt1[tid];
        hbuf[tid] = gelu_exact(v);
    }
    __syncthreads();
    #pragma unroll
    for (int i = 0; i < 3; ++i){
        int d = tid + i * 256;
        const float* wrow = se_w2 + (size_t)d * D4_N;
        float acc = se_b2[d];
        for (int j = 0; j < D4_N; ++j) acc += hbuf[j] * wrow[j];
        rowS[d] = (_Float16)acc;
    }
}

// ---------------- Kernel 2: TDM-staged QKV GEMM + 4-token attention -----
// block = 256 threads (8 waves), 4 elements (one M=16 WMMA tile row).
// N processed in 9 chunks of 256 cols; B tiles (256x32 f16) double-buffered
// in LDS via tensor_load_to_lds, A panel (16x768) staged once via TDM.
__global__ __launch_bounds__(256) void k_qkv_attn(
    const _Float16* __restrict__ A1,     // [4B][768] feats, f16
    const _Float16* __restrict__ W1h,    // [2304][768] in_proj_w, f16 row-major
    const float*    __restrict__ in_b,   // [2304]
    _Float16*       __restrict__ A2)     // [B][768] token-weighted attn out
{
    __shared__ __align__(32) _Float16 qkv[16][TD_N];     // 72 KB
    __shared__ __align__(32) _Float16 Ast[16][D_N];      // 24 KB
    __shared__ __align__(32) _Float16 Bst[2][256][32];   // 32 KB
    __shared__ float scores[4][8][4][4];
    __shared__ float att[4][8][4][4];

    const int e0   = blockIdx.x * 4;
    const int m0   = e0 * 4;
    const int wave = threadIdx.x >> 5;
    const int lane = threadIdx.x & 31;
    const int l16  = lane & 15;
    const int hi   = lane >> 4;
    const int sA = hi * 16;
    const int tB = hi * 16;
    const int NSTEP = 9 * 24;                 // chunks x K-steps

    // prologue: wave 0 kicks off A panel + first B tile
    if (threadIdx.x < 32){
        tdm_load_2d_f16((unsigned int)(uintptr_t)&Ast[0][0],
                        A1 + (size_t)m0 * D_N, 16 * D_N, 1, 16 * D_N);
        tdm_load_2d_f16((unsigned int)(uintptr_t)&Bst[0][0][0],
                        W1h, 32, 256, D_N);
    }

    int s = 0;
    for (int c = 0; c < 9; ++c){
        v8f acc0 = (v8f)0.f, acc1 = (v8f)0.f;
        for (int ks = 0; ks < 24; ++ks, ++s){
            if (threadIdx.x < 32){
                if (s + 1 < NSTEP){
                    int cn = (s + 1) / 24, kn = ((s + 1) % 24) * 32;
                    tdm_load_2d_f16((unsigned int)(uintptr_t)&Bst[(s + 1) & 1][0][0],
                                    W1h + (size_t)(cn * 256) * D_N + kn, 32, 256, D_N);
                    __builtin_amdgcn_s_wait_tensorcnt(1);
                } else {
                    __builtin_amdgcn_s_wait_tensorcnt(0);
                }
            }
            __syncthreads();                 // buffer p ready for everyone
            const int p  = s & 1;
            const int kb = ks * 32;
            v8h a0 = *(const v8h*)&Ast[l16][kb + sA];
            v8h a1 = *(const v8h*)&Ast[l16][kb + 8 + sA];
            v16h a = __builtin_shufflevector(a0, a1,
                     0,1,2,3,4,5,6,7,8,9,10,11,12,13,14,15);
            v16h b0 = *(const v16h*)&Bst[p][wave * 32 + l16][tB];
            v16h b1 = *(const v16h*)&Bst[p][wave * 32 + 16 + l16][tB];
            acc0 = __builtin_amdgcn_wmma_f32_16x16x32_f16(
                false, a, false, b0, (short)0, acc0, false, false);
            acc1 = __builtin_amdgcn_wmma_f32_16x16x32_f16(
                false, a, false, b1, (short)0, acc1, false, false);
            __syncthreads();                 // done with buffer p before refill
        }
        // scatter chunk results (+bias) into qkv LDS
        #pragma unroll
        for (int t = 0; t < 2; ++t){
            int n = c * 256 + wave * 32 + t * 16 + l16;
            float bias = in_b[n];
            #pragma unroll
            for (int r = 0; r < 8; ++r){
                float v = ((t == 0) ? acc0[r] : acc1[r]) + bias;
                qkv[hi * 8 + r][n] = (_Float16)v;
            }
        }
    }
    __syncthreads();

    // scores: 4 elem x 8 heads x 4x4 = 512 dots of length 96; 2 per thread
    const float rscale = 0.10206207261596576f;   // 1/sqrt(96)
    #pragma unroll
    for (int k2 = 0; k2 < 2; ++k2){
        int sidx = threadIdx.x * 2 + k2;
        int e  = sidx >> 7;
        int h  = (sidx >> 4) & 7;
        int tq = (sidx >> 2) & 3;
        int tu = sidx & 3;
        const _Float16* qp = &qkv[e * 4 + tq][h * 96];
        const _Float16* kp = &qkv[e * 4 + tu][D_N + h * 96];
        float sum = 0.f;
        for (int d = 0; d < 96; ++d) sum += (float)qp[d] * (float)kp[d];
        scores[e][h][tq][tu] = sum * rscale;
    }
    __syncthreads();

    if (threadIdx.x < 128){
        int e  = threadIdx.x >> 5;
        int h  = (threadIdx.x >> 2) & 7;
        int tq = threadIdx.x & 3;
        float s0 = scores[e][h][tq][0], s1 = scores[e][h][tq][1];
        float s2 = scores[e][h][tq][2], s3 = scores[e][h][tq][3];
        float mx = fmaxf(fmaxf(s0, s1), fmaxf(s2, s3));
        float e_0 = expf(s0 - mx), e_1 = expf(s1 - mx);
        float e_2 = expf(s2 - mx), e_3 = expf(s3 - mx);
        float inv = 1.f / (e_0 + e_1 + e_2 + e_3);
        att[e][h][tq][0] = e_0 * inv; att[e][h][tq][1] = e_1 * inv;
        att[e][h][tq][2] = e_2 * inv; att[e][h][tq][3] = e_3 * inv;
    }
    __syncthreads();

    const float w0 = 0.4f, w1 = 0.3f, w2 = 0.2f, w3 = 0.1f;
    #pragma unroll
    for (int c = 0; c < 12; ++c){
        int idx = threadIdx.x + c * 256;     // 0 .. 3071
        int e  = idx / D_N;
        int dh = idx % D_N;
        int h  = dh / 96;
        int dd = dh % 96;
        float val = 0.f;
        #pragma unroll
        for (int tu = 0; tu < 4; ++tu){
            float coef = w0 * att[e][h][0][tu] + w1 * att[e][h][1][tu]
                       + w2 * att[e][h][2][tu] + w3 * att[e][h][3][tu];
            val += coef * (float)qkv[e * 4 + tu][2 * D_N + h * 96 + dd];
        }
        A2[(size_t)(e0 + e) * D_N + dh] = (_Float16)val;
    }
}

// ---------------- Kernel 3: out_proj GEMM + scale/NaN/L2-norm epilogue ---
__global__ __launch_bounds__(256) void k_outproj(
    const _Float16* __restrict__ A2,     // [B][768] f16
    const _Float16* __restrict__ W2h,    // [768][768] out_proj_w f16 row-major
    const float*    __restrict__ ob,     // [768]
    const float*    __restrict__ sc,     // [B]
    float*          __restrict__ out)    // [B][768]
{
    __shared__ float obuf[16][D_N];      // 48 KB
    __shared__ float red[16][16];
    __shared__ float invn[16];

    const int m0   = blockIdx.x * 16;
    const int wave = threadIdx.x >> 5;
    const int lane = threadIdx.x & 31;
    const int l16  = lane & 15;
    const int hi   = lane >> 4;
    const int nbase = wave * 96;

    v8f acc[6];
    #pragma unroll
    for (int t = 0; t < 6; ++t) acc[t] = (v8f)0.f;

    const _Float16* arow = A2 + (size_t)(m0 + l16) * D_N;
    const int sA = hi * 16, tB = hi * 16;

    for (int kb = 0; kb < D_N; kb += 32){
        v8h a0 = *(const v8h*)(arow + kb + sA);
        v8h a1 = *(const v8h*)(arow + kb + 8 + sA);
        v16h a = __builtin_shufflevector(a0, a1,
                 0,1,2,3,4,5,6,7,8,9,10,11,12,13,14,15);
        #pragma unroll
        for (int t = 0; t < 6; ++t){
            const _Float16* bp = W2h + (size_t)(nbase + t * 16 + l16) * D_N + kb + tB;
            v16h bfrag = *(const v16h*)bp;
            acc[t] = __builtin_amdgcn_wmma_f32_16x16x32_f16(
                false, a, false, bfrag, (short)0, acc[t], false, false);
        }
    }

    #pragma unroll
    for (int t = 0; t < 6; ++t){
        int n = nbase + t * 16 + l16;
        float bias = ob[n];
        #pragma unroll
        for (int r = 0; r < 8; ++r){
            int m = hi * 8 + r;
            float v = (acc[t][r] + bias) * sc[m0 + m];
            if (v != v) v = 0.f;         // NaN -> 0
            obuf[m][n] = v;
        }
    }
    __syncthreads();

    {
        int r = threadIdx.x >> 4, c = threadIdx.x & 15;
        float s = 0.f;
        for (int i = 0; i < 48; ++i){ float v = obuf[r][c * 48 + i]; s += v * v; }
        red[r][c] = s;
    }
    __syncthreads();
    if (threadIdx.x < 16){
        float s = 0.f;
        for (int c = 0; c < 16; ++c) s += red[threadIdx.x][c];
        invn[threadIdx.x] = 1.f / fmaxf(sqrtf(s), 1e-12f);
    }
    __syncthreads();

    #pragma unroll
    for (int c = 0; c < 48; ++c){
        int idx = threadIdx.x + c * 256;
        int r = idx / D_N, n = idx % D_N;
        out[(size_t)(m0 + r) * D_N + n] = obuf[r][n] * invn[r];
    }
}

// ------------------------------------------------------------------------
extern "C" void kernel_launch(void* const* d_in, const int* in_sizes, int n_in,
                              void* d_out, int out_size, void* d_ws, size_t ws_size,
                              hipStream_t stream)
{
    const float* number     = (const float*)d_in[0];
    const float* boundaries = (const float*)d_in[1];
    const float* mag_emb    = (const float*)d_in[2];
    const float* se_w1  = (const float*)d_in[3];
    const float* se_b1  = (const float*)d_in[4];
    const float* se_g1  = (const float*)d_in[5];
    const float* se_bt1 = (const float*)d_in[6];
    const float* se_w2  = (const float*)d_in[7];
    const float* se_b2  = (const float*)d_in[8];
    const float* ne_w   = (const float*)d_in[9];
    const float* ne_b   = (const float*)d_in[10];
    const float* ne_g   = (const float*)d_in[11];
    const float* ne_bt  = (const float*)d_in[12];
    const float* ce_w   = (const float*)d_in[13];
    const float* ce_b   = (const float*)d_in[14];
    const float* ce_g   = (const float*)d_in[15];
    const float* ce_bt  = (const float*)d_in[16];
    const float* in_w   = (const float*)d_in[17];
    const float* in_b   = (const float*)d_in[18];
    const float* out_w  = (const float*)d_in[19];
    const float* out_b  = (const float*)d_in[20];
    const float* mag_scale   = (const float*)d_in[21];
    const float* temperature = (const float*)d_in[22];

    char* ws = (char*)d_ws;
    size_t off = 0;
    auto walloc = [&](size_t bytes) -> char* {
        char* p = ws + off;
        off = (off + bytes + 255) & ~(size_t)255;
        return p;
    };
    float*     maxv = (float*)    walloc(2 * sizeof(float));
    float*     scb  = (float*)    walloc((size_t)B_N * sizeof(float));
    _Float16*  W1h  = (_Float16*) walloc((size_t)TD_N * D_N * sizeof(_Float16));
    _Float16*  W2h  = (_Float16*) walloc((size_t)D_N * D_N * sizeof(_Float16));
    _Float16*  A1   = (_Float16*) walloc((size_t)B_N * 4 * D_N * sizeof(_Float16));
    _Float16*  A2   = (_Float16*) walloc((size_t)B_N * D_N * sizeof(_Float16));

    k_maxred<<<1, 256, 0, stream>>>(number, maxv);
    k_cvt<<<1024, 256, 0, stream>>>(in_w,  W1h, TD_N * D_N);
    k_cvt<<<512,  256, 0, stream>>>(out_w, W2h, D_N * D_N);
    k_features<<<B_N, 256, 0, stream>>>(
        number, boundaries, mag_emb,
        se_w1, se_b1, se_g1, se_bt1, se_w2, se_b2,
        ne_w, ne_b, ne_g, ne_bt, ce_w, ce_b, ce_g, ce_bt,
        mag_scale, temperature, maxv, A1, scb);
    k_qkv_attn<<<B_N / 4, 256, 0, stream>>>(A1, W1h, in_b, A2);
    k_outproj<<<B_N / 16, 256, 0, stream>>>(A2, W2h, out_b, scb, (float*)d_out);
}
